// Generator_45492293599886
// MI455X (gfx1250) — matrix operand order repaired and checked by reading, not compiled, not run on hardware
//
#include <hip/hip_runtime.h>

// AJ-RNN fused forward for MI455X (gfx1250, wave32, WMMA bf16 16x16x32).
// Persistent kernel: 32 WGs x 256 threads, weight-stationary in LDS,
// device-wide sense-reversing barrier between recurrence phases.

typedef __attribute__((ext_vector_type(16))) __bf16 bf16x16;
typedef __attribute__((ext_vector_type(8)))  float  f32x8;

#define B_   128
#define T_   256
#define D_   64
#define H_   512
#define C_   10
#define KC   576      // D_ + H_  (concatenated GEMM K)
#define NWG  32
#define NTH  256
#define NKT  18       // KC / 32

// Persistent device state. gBar generation counter is monotonic across
// launches (sense-reversing barrier snapshots it), counter self-resets to 0,
// and gXb / partials are fully rewritten before every read each launch.
__device__ __attribute__((aligned(16))) __bf16 gXb[2][B_ * KC];
__device__ unsigned gBar[2];
__device__ float    gRegPart[NWG];
__device__ float    gLpPart[NWG];

__device__ __forceinline__ float sigf(float x) { return 1.0f / (1.0f + __expf(-x)); }

// A fragment (16x32 bf16, row-major source [rows, KC]):
// lane: M = m0 + (lane&15); K runs k0+16*(lane>>4)+[0..15].
__device__ __forceinline__ bf16x16 ldA(const __bf16* __restrict__ base, int m0, int k0, int lane) {
  const __bf16* rp = base + (m0 + (lane & 15)) * KC + k0 + ((lane >> 4) << 4);
  union { uint4 u[2]; bf16x16 v; } t;
  t.u[0] = *(const uint4*)(rp);
  t.u[1] = *(const uint4*)(rp + 8);
  return t.v;
}

// B fragment (32x16 bf16) pre-swizzled in LDS: frag*512 + lane*16 bf16.
__device__ __forceinline__ bf16x16 ldB(const __bf16* p, int frag, int lane) {
  const __bf16* q = p + frag * 512 + lane * 16;
  union { uint4 u[2]; bf16x16 v; } t;
  t.u[0] = *(const uint4*)(q);
  t.u[1] = *(const uint4*)(q + 8);
  return t.v;
}

__device__ __forceinline__ f32x8 wmma_bf16(bf16x16 a, bf16x16 b, f32x8 c) {
  return __builtin_amdgcn_wmma_f32_16x16x32_bf16(false, a, false, b, (short)0, c, false, false);
}

// Device-wide sense-reversing barrier. Counter self-resets; gen is monotonic.
__device__ __forceinline__ void gbar() {
  __syncthreads();
  if (threadIdx.x == 0) {
    __threadfence();
    unsigned g = __atomic_load_n(&gBar[1], __ATOMIC_ACQUIRE);
    unsigned v = atomicAdd(&gBar[0], 1u);
    if (v == NWG - 1u) {
      __atomic_store_n(&gBar[0], 0u, __ATOMIC_RELAXED);
      __threadfence();
      atomicAdd(&gBar[1], 1u);
    } else {
      while (__atomic_load_n(&gBar[1], __ATOMIC_ACQUIRE) == g)
        __builtin_amdgcn_s_sleep(1);
    }
    __threadfence();
  }
  __syncthreads();
}

__global__ void __launch_bounds__(NTH, 1)
ajrnn_fused(const float* __restrict__ x,       // [B,T,D]
            const float* __restrict__ ptgt,    // [B,T-1,D]
            const float* __restrict__ pmask,   // [B,T-1,D]
            const float* __restrict__ ltgt,    // [B,C] one-hot
            const float* __restrict__ Wp,      // [H,D]
            const float* __restrict__ biasP,   // [D]
            const float* __restrict__ kern,    // [D,4H]
            const float* __restrict__ rkern,   // [H,4H]
            const float* __restrict__ lbias,   // [4H]
            const float* __restrict__ dW,      // [H,C]
            const float* __restrict__ db,      // [C]
            float* __restrict__ out)           // [B]
{
  __shared__ __attribute__((aligned(16))) __bf16 shW[72 * 512];  // LSTM weight slice, frag layout
  __shared__ __attribute__((aligned(16))) __bf16 shP[16 * 512];  // projection W slice, frag layout
  __shared__ float sred[8];

  const int wg     = blockIdx.x;
  const int tid    = threadIdx.x;
  const int wave   = tid >> 5;
  const int lane   = tid & 31;
  const int lanelo = lane & 15;
  const int lhalf  = lane >> 4;

  // ---------- init: swizzle this WG's LSTM weight slice (4 gates x 16 cols) ----------
  for (int f = wave; f < 72; f += 8) {
    const int kt = f >> 2, q = f & 3;
    const int col = q * H_ + wg * 16 + lanelo;
#pragma unroll
    for (int d = 0; d < 8; ++d) {
      const int K0 = kt * 32 + (lhalf << 4) + d * 2;  // even
      float a0 = (K0 < D_) ? kern[K0 * (4 * H_) + col] : rkern[(K0 - D_) * (4 * H_) + col];
      float a1 = (K0 + 1 < D_) ? kern[(K0 + 1) * (4 * H_) + col]
                               : rkern[(K0 + 1 - D_) * (4 * H_) + col];
      shW[f * 512 + lane * 16 + d * 2]     = (__bf16)a0;
      shW[f * 512 + lane * 16 + d * 2 + 1] = (__bf16)a1;
    }
  }

  // P-tile assignment: 8 M-tiles x 4 N-tiles of the [128,64] projection
  const int pm0 = (wg >> 2) * 16;
  const int pn0 = (wg & 3) * 16;

  // projection-W slice (K=512, 16 cols) into LDS frag layout
  for (int f = wave; f < 16; f += 8) {
    const int col = pn0 + lanelo;
#pragma unroll
    for (int d = 0; d < 8; ++d) {
      const int K0 = f * 32 + (lhalf << 4) + d * 2;
      shP[f * 512 + lane * 16 + d * 2]     = (__bf16)Wp[K0 * D_ + col];
      shP[f * 512 + lane * 16 + d * 2 + 1] = (__bf16)Wp[(K0 + 1) * D_ + col];
    }
  }

  // ---------- init gXb[0]: x at t=0 (raw, no imputation) and h0 = 0 ----------
  {
    const int r = tid >> 4, c = tid & 15;
    gXb[0][(pm0 + r) * KC + pn0 + c] = (__bf16)x[((pm0 + r) * T_ + 0) * D_ + pn0 + c];
    for (int i = tid; i < 128 * 16; i += NTH) {
      const int m = i >> 4, cc = 64 + wg * 16 + (i & 15);
      gXb[0][m * KC + cc] = (__bf16)0.0f;
    }
  }

  // ---------- L2 regularization partial (deterministic tree) ----------
  {
    float rs = 0.f;
    const int gid = wg * NTH + tid, gstr = NWG * NTH;
    const float* arrs[7] = {Wp, biasP, kern, rkern, lbias, dW, db};
    const int    lens[7] = {H_ * D_, D_, D_ * 4 * H_, H_ * 4 * H_, 4 * H_, H_ * C_, C_};
    for (int a2 = 0; a2 < 7; ++a2) {
      const float* p = arrs[a2];
      const int n = lens[a2];
      for (int i = gid; i < n; i += gstr) { float v = p[i]; rs += v * v; }
    }
#pragma unroll
    for (int mK = 16; mK; mK >>= 1) rs += __shfl_xor(rs, mK, 32);
    if (lane == 0) sred[wave] = rs;
    __syncthreads();
    if (tid == 0) {
      float s = 0.f;
      for (int i2 = 0; i2 < 8; ++i2) s += sred[i2];
      gRegPart[wg] = 0.5f * s;
    }
  }

  // per-thread gate biases + persistent cell state (8 f32 per lane)
  const float lb0 = lbias[0 * H_ + wg * 16 + lanelo];
  const float lb1 = lbias[1 * H_ + wg * 16 + lanelo];
  const float lb2 = lbias[2 * H_ + wg * 16 + lanelo];
  const float lb3 = lbias[3 * H_ + wg * 16 + lanelo];
  const float pbias = biasP[pn0 + lanelo];
  float cst[8];
#pragma unroll
  for (int r = 0; r < 8; ++r) cst[r] = 0.f;

  __syncthreads();
  gbar();  // gXb[0] complete everywhere

  const int m0w   = wave * 16;                 // this wave's M-tile in the cell GEMM
  const int hcol  = 64 + wg * 16 + lanelo;     // h column this lane writes
  const int mbase = m0w + (lhalf << 3);

  // One LSTM cell step: z = Xin @ [kernel;rkernel], gates, write h to Xout.
  auto lstm_phase = [&](const __bf16* Xin, __bf16* Xout) {
    f32x8 zi = {0, 0, 0, 0, 0, 0, 0, 0};
    f32x8 zf = zi, zg = zi, zo = zi;
    for (int kt = 0; kt < NKT; ++kt) {
      bf16x16 a = ldA(Xin, m0w, kt * 32, lane);
      zi = wmma_bf16(a, ldB(shW, kt * 4 + 0, lane), zi);
      zf = wmma_bf16(a, ldB(shW, kt * 4 + 1, lane), zf);
      zg = wmma_bf16(a, ldB(shW, kt * 4 + 2, lane), zg);
      zo = wmma_bf16(a, ldB(shW, kt * 4 + 3, lane), zo);
    }
#pragma unroll
    for (int r = 0; r < 8; ++r) {
      float c = sigf(zf[r] + lb1) * cst[r] + sigf(zi[r] + lb0) * tanhf(zg[r] + lb2);
      cst[r] = c;
      float h = sigf(zo[r] + lb3) * tanhf(c);
      Xout[(mbase + r) * KC + hcol] = (__bf16)h;
    }
  };

  // step 0: raw input -> h1 (into buffer 1)
  lstm_phase(gXb[0], gXb[1]);
  gbar();

  float lpAcc = 0.f;  // running masked-MSE partial (wave 0 only)

  for (int s = 1; s < T_; ++s) {
    const int bi = s & 1;  // buffer holding h_s (and receiving x_in for step s)
    if (wave == 0) {
      // P = h_s @ Wp + biasP  (one 16x16 tile per WG), dual use:
      // prediction-branch loss row s-1, and imputation for x[:, s, :].
      f32x8 p = {0, 0, 0, 0, 0, 0, 0, 0};
      for (int kt = 0; kt < 16; ++kt) {
        bf16x16 a = ldA(gXb[bi], pm0, 64 + kt * 32, lane);
        p = wmma_bf16(a, ldB(shP, kt, lane), p);
      }
      const int colD = pn0 + lanelo;
#pragma unroll
      for (int r = 0; r < 8; ++r) {
        const int m = pm0 + (lhalf << 3) + r;
        float pv = p[r] + pbias;
        float xv = x[(m * T_ + s) * D_ + colD];
        float xin = (xv == 128.0f) ? pv : xv;
        gXb[bi][m * KC + colD] = (__bf16)xin;
        const int idx = (m * (T_ - 1) + (s - 1)) * D_ + colD;
        float dd = (ptgt[idx] - pv) * pmask[idx];
        lpAcc += dd * dd;
      }
    }
    gbar();                               // x_in visible device-wide
    lstm_phase(gXb[bi], gXb[1 - bi]);     // h_{s+1} into the other buffer
    gbar();                               // h_{s+1} visible device-wide
  }
  // final h_256 lives in gXb[0] (s=255 wrote buffer 1-(255&1)=0)

  if (wave == 0) {
    float v = lpAcc;
#pragma unroll
    for (int mK = 16; mK; mK >>= 1) v += __shfl_xor(v, mK, 32);
    if (lane == 0) gLpPart[wg] = v;
  }
  gbar();

  // ---------- epilogue: classification + combine (WG 0, one thread per batch row) ----------
  if (wg == 0 && tid < B_) {
    float regS = 0.f, lpS = 0.f;
    for (int i = 0; i < NWG; ++i) { regS += gRegPart[i]; lpS += gLpPart[i]; }
    float logits[C_];
#pragma unroll
    for (int c = 0; c < C_; ++c) logits[c] = db[c];
    const __bf16* hrow = &gXb[0][tid * KC + 64];
    for (int k = 0; k < H_; ++k) {
      float hv = (float)hrow[k];
#pragma unroll
      for (int c = 0; c < C_; ++c) logits[c] += hv * dW[k * C_ + c];
    }
    float mx = logits[0];
#pragma unroll
    for (int c = 1; c < C_; ++c) mx = fmaxf(mx, logits[c]);
    float se = 0.f;
#pragma unroll
    for (int c = 0; c < C_; ++c) se += __expf(logits[c] - mx);
    const float lse = mx + __logf(se);
    float lc = 0.f;
#pragma unroll
    for (int c = 0; c < C_; ++c) lc += ltgt[tid * C_ + c] * (logits[c] - lse);
    const float pscale = 1.0f / ((float)B_ * (float)(T_ - 1) * (float)D_ * (float)B_);
    out[tid] = -lc + lpS * pscale + 1e-4f * regS;
  }
}

extern "C" void kernel_launch(void* const* d_in, const int* in_sizes, int n_in,
                              void* d_out, int out_size, void* d_ws, size_t ws_size,
                              hipStream_t stream) {
  (void)in_sizes; (void)n_in; (void)out_size; (void)d_ws; (void)ws_size;
  ajrnn_fused<<<NWG, NTH, 0, stream>>>(
      (const float*)d_in[0],  // input
      (const float*)d_in[1],  // prediction_target
      (const float*)d_in[2],  // mask
      (const float*)d_in[3],  // label_target
      (const float*)d_in[4],  // W
      (const float*)d_in[5],  // bias
      (const float*)d_in[6],  // kernel
      (const float*)d_in[7],  // rkernel
      (const float*)d_in[8],  // lstm_bias
      (const float*)d_in[9],  // dense_W
      (const float*)d_in[10], // dense_b
      (float*)d_out);
}
